// RTMDetLoss_42305427865903
// MI455X (gfx1250) — compile-verified, matrix operand order checked by reading, not run on hardware
//
#include <hip/hip_runtime.h>
#include <hip/hip_bf16.h>
#include <stdint.h>

typedef __attribute__((ext_vector_type(16))) _Float16 v16h;
typedef __attribute__((ext_vector_type(8)))  float    v8f;

#define NCLS  80
#define TOPKC 13
#define BIGC  100000000.0f
#define EPSC  1e-7f
#define CRAD  2.5f
// clip(p,1e-6,1-1e-6):  -log(1-p) in [1.0000005e-6, 13.8155106]
#define SP_LO 1.0000005e-6f
#define SP_HI 13.8155106f

// ---------------------------------------------------------------- kernel 0
__global__ __launch_bounds__(256) void k_init(unsigned long long* __restrict__ best,
                                              float* __restrict__ acc,
                                              unsigned* __restrict__ cnt, int N) {
  int i = blockIdx.x * blockDim.x + threadIdx.x;
  if (i < N) best[i] = ~0ULL;
  if (i == 0) { acc[0] = 0.f; acc[1] = 0.f; cnt[0] = 0u; cnt[1] = 0u; }
}

// ---------------------------------------------------------------- kernel 1
// Per wave: 16 anchors. all_neg via WMMA row-reduction over classes
// (A = clamp(softplus(x)) tile in f16, B = ones).  -log(1-sigmoid(x)) ==
// softplus(x); the reference's p-clip becomes a value clamp => no division.
// cand_any via LDS-staged gt boxes (CDNA5 async copy), 2 lanes per anchor.
__global__ __launch_bounds__(256) void k_prep(const float* __restrict__ logits,
                                              const float* __restrict__ points,
                                              const float* __restrict__ strides,
                                              const float4* __restrict__ gt_boxes,
                                              int N, int M,
                                              float* __restrict__ all_neg,
                                              int* __restrict__ cand_any) {
  __shared__ float4 sgt[256];
  for (int t = threadIdx.x; t < M && t < 256; t += blockDim.x) {
    unsigned lds = (unsigned)(size_t)(&sgt[t]);
    unsigned long long ga = (unsigned long long)(size_t)(gt_boxes + t);
    asm volatile("global_load_async_to_lds_b128 %0, %1, off"
                 :: "v"(lds), "v"(ga) : "memory");
  }
  asm volatile("s_wait_asynccnt 0" ::: "memory");
  __syncthreads();

  const int tid   = threadIdx.x;
  const int lane  = tid & 31;
  const int wave  = tid >> 5;
  const int base  = blockIdx.x * 128 + wave * 16;   // 16 anchors per wave
  const int mrow  = lane & 15;
  const int arow  = (base + mrow < N) ? (base + mrow) : (N - 1);
  const int khalf = (lane >> 4) * 16;               // contiguous K-half per lane half (this code's order)

  v16h bones;
#pragma unroll
  for (int j = 0; j < 16; ++j) bones[j] = (_Float16)1.0f;

  v8f acc = {};
#pragma unroll
  for (int s = 0; s < 3; ++s) {                     // K = 96 (80 classes + pad)
    v16h a;
    const int cbase = s * 32 + khalf;
#pragma unroll
    for (int j = 0; j < 16; ++j) {
      const int c = cbase + j;
      float v = 0.0f;
      if (c < NCLS) {
        float x = logits[(size_t)arow * NCLS + c];
        v = log1pf(__expf(x));                      // softplus == -log(1-p)
        v = fminf(fmaxf(v, SP_LO), SP_HI);          // == p-clip
      }
      a[j] = (_Float16)v;
    }
    acc = __builtin_amdgcn_wmma_f32_16x16x32_f16(false, a, false, bones,
                                                 (short)0, acc, false, false);
  }
  // D layout: VGPR r, lanes 0-15 -> (M=r, N=lane); lanes 16-31 -> (M=8+r).
  if (lane == 0) {
#pragma unroll
    for (int r = 0; r < 8; ++r) { int a0 = base + r;     if (a0 < N) all_neg[a0] = acc[r]; }
  } else if (lane == 16) {
#pragma unroll
    for (int r = 0; r < 8; ++r) { int a0 = base + 8 + r; if (a0 < N) all_neg[a0] = acc[r]; }
  }

  // cand_any: lane l and l+16 split the gt range for anchor base+(l&15).
  const int aidx = base + mrow;
  float px = 0.f, py = 0.f, st = 0.f;
  if (aidx < N) { px = points[2 * aidx]; py = points[2 * aidx + 1]; st = strides[aidx]; }
  const float r = CRAD * st;
  const int half = lane >> 4;
  int any = 0;
  for (int m = half; m < M; m += 2) {
    float4 g = sgt[m];
    int inb = (px > g.x) & (py > g.y) & (px < g.z) & (py < g.w);
    float gcx = (g.x + g.z) * 0.5f, gcy = (g.y + g.w) * 0.5f;
    int inc = (px > gcx - r) & (py > gcy - r) & (px < gcx + r) & (py < gcy + r);
    any |= (inb | inc);
  }
  any |= __shfl_xor(any, 16, 32);
  if (half == 0 && aidx < N) cand_any[aidx] = any;
}

// ---------------------------------------------------------------- kernel 2
// One workgroup per GT column. Full cost + iou columns live in LDS (2*4*N B
// <= 320KB WGP LDS). 13 argmax passes -> dyn_k; dyn_k argmin passes -> commit
// matches via packed atomicMin (min-cost gt per anchor, tie -> lowest m).
// cls gather term: log(1-p)-log(p) == -x (clip => saturate) -- division-free.
__global__ __launch_bounds__(1024) void k_cols(const float* __restrict__ logits,
                                               const float4* __restrict__ pboxes,
                                               const float2* __restrict__ points,
                                               const float* __restrict__ strides,
                                               const float4* __restrict__ gt_boxes,
                                               const int* __restrict__ gt_labels,
                                               const float* __restrict__ all_neg,
                                               const int* __restrict__ cand_any,
                                               unsigned long long* __restrict__ best,
                                               int N) {
  extern __shared__ float smem[];
  float* costL = smem;
  float* iouL  = smem + N;
  __shared__ float rv[1024];
  __shared__ int   ri[1024];

  const int m   = blockIdx.x;
  const int tid = threadIdx.x;
  const int bs  = blockDim.x;

  const float4 g = gt_boxes[m];
  const int label = gt_labels[m];
  const float areaB = (g.z - g.x) * (g.w - g.y);
  const float gcx = (g.x + g.z) * 0.5f, gcy = (g.y + g.w) * 0.5f;

  for (int i = tid; i < N; i += bs) {
    float4 pb = pboxes[i];
    float2 pt = points[i];
    float st  = strides[i];
    float lx = fmaxf(pb.x, g.x), ly = fmaxf(pb.y, g.y);
    float rx = fminf(pb.z, g.z), ry = fminf(pb.w, g.w);
    float inter = fmaxf(rx - lx, 0.f) * fmaxf(ry - ly, 0.f);
    float areaA = (pb.z - pb.x) * (pb.w - pb.y);
    float uni = areaA + areaB - inter;
    float iou = inter / fmaxf(uni, EPSC);
    int inb = (pt.x > g.x) & (pt.y > g.y) & (pt.x < g.z) & (pt.y < g.w);
    float r = CRAD * st;
    int inc = (pt.x > gcx - r) & (pt.y > gcy - r) & (pt.x < gcx + r) & (pt.y < gcy + r);
    int cand = inb | inc;

    float x = logits[(size_t)i * NCLS + label];
    {
      int ip = i + bs; if (ip >= N) ip = i;
      __builtin_prefetch(&logits[(size_t)ip * NCLS + label], 0, 1);  // global_prefetch_b8
    }
    float q = fminf(fmaxf(-x, -SP_HI), SP_HI);      // log(1-p)-log(p), p clipped
    float cost = all_neg[i] + q - 3.0f * __logf(iou + EPSC);
    costL[i] = cand ? cost : BIGC;
    iouL[i]  = cand_any[i] ? iou : 0.0f;
  }
  __syncthreads();

  // dyn_k = clamp(int(sum of top-13 ious), 1, 13)
  float sum13 = 0.0f;
  for (int k = 0; k < TOPKC; ++k) {
    float bv = -2.0f; int bi = 0x7fffffff;
    for (int i = tid; i < N; i += bs) {
      float v = iouL[i];
      if (v > bv || (v == bv && i < bi)) { bv = v; bi = i; }
    }
    rv[tid] = bv; ri[tid] = bi; __syncthreads();
    for (int s = bs >> 1; s > 0; s >>= 1) {
      if (tid < s) {
        if (rv[tid + s] > rv[tid] ||
            (rv[tid + s] == rv[tid] && ri[tid + s] < ri[tid])) {
          rv[tid] = rv[tid + s]; ri[tid] = ri[tid + s];
        }
      }
      __syncthreads();
    }
    float v = rv[0]; int idx = ri[0];
    sum13 += fmaxf(v, 0.0f);
    if (tid == 0) iouL[idx] = -1.0f;    // exclude
    __syncthreads();
  }
  int dyn_k = (int)sum13; if (dyn_k < 1) dyn_k = 1; if (dyn_k > TOPKC) dyn_k = TOPKC;

  // top-dyn_k lowest costs -> matching; commit packed (sortable_cost, m).
  for (int k = 0; k < dyn_k; ++k) {
    float bv = 3.4e38f; int bi = 0x7fffffff;
    for (int i = tid; i < N; i += bs) {
      float v = costL[i];
      if (v < bv || (v == bv && i < bi)) { bv = v; bi = i; }
    }
    rv[tid] = bv; ri[tid] = bi; __syncthreads();
    for (int s = bs >> 1; s > 0; s >>= 1) {
      if (tid < s) {
        if (rv[tid + s] < rv[tid] ||
            (rv[tid + s] == rv[tid] && ri[tid + s] < ri[tid])) {
          rv[tid] = rv[tid + s]; ri[tid] = ri[tid + s];
        }
      }
      __syncthreads();
    }
    float v = rv[0]; int idx = ri[0];
    __syncthreads();
    if (v >= BIGC) break;               // uniform: remaining mins only grow
    if (tid == 0) {
      unsigned ub = __float_as_uint(v);
      unsigned key = (ub & 0x80000000u) ? ~ub : (ub | 0x80000000u);
      unsigned long long pk = ((unsigned long long)key << 32) | (unsigned)m;
      atomicMin(best + idx, pk);
      costL[idx] = BIGC;                // exclude
    }
    __syncthreads();
  }
}

// ---------------------------------------------------------------- kernel 3
__global__ __launch_bounds__(256) void k_loss(const float* __restrict__ logits,
                                              const float4* __restrict__ pboxes,
                                              const float4* __restrict__ gt_boxes,
                                              const int* __restrict__ gt_labels,
                                              const unsigned long long* __restrict__ best,
                                              int N,
                                              float* __restrict__ acc,
                                              unsigned* __restrict__ cnt) {
  const int i = blockIdx.x * blockDim.x + threadIdx.x;
  float s_cls = 0.f, s_box = 0.f;
  unsigned c_pos = 0, c_match = 0;
  if (i < N) {
    unsigned long long pk = best[i];
    bool matched = (pk != ~0ULL);
    int label = -1; float pos_iou = 0.f;
    if (matched) {
      int m = (int)(unsigned)pk;
      float4 gb = gt_boxes[m]; float4 a = pboxes[i];
      label = gt_labels[m];
      float lx = fmaxf(a.x, gb.x), ly = fmaxf(a.y, gb.y);
      float rx = fminf(a.z, gb.z), ry = fminf(a.w, gb.w);
      float inter = fmaxf(rx - lx, 0.f) * fmaxf(ry - ly, 0.f);
      float areaA = (a.z - a.x) * (a.w - a.y);
      float areaB = (gb.z - gb.x) * (gb.w - gb.y);
      float uni = areaA + areaB - inter;
      float iou = inter / fmaxf(uni, EPSC);
      pos_iou = iou;
      float ex = fminf(a.x, gb.x), ey = fminf(a.y, gb.y);
      float eX = fmaxf(a.z, gb.z), eY = fmaxf(a.w, gb.w);
      float enc = fmaxf(eX - ex, 0.f) * fmaxf(eY - ey, 0.f);
      float giou = iou - (enc - uni) / fmaxf(enc, EPSC);
      s_box = 1.0f - giou;
      c_match = 1u;
      if (pos_iou > 0.f) c_pos = 1u;
    }
    for (int c = 0; c < NCLS; ++c) {          // QFL over all anchors
      float x = logits[(size_t)i * NCLS + c];
      float p = __builtin_amdgcn_rcpf(1.0f + __expf(-x));   // fast sigmoid
      float t = (c == label) ? pos_iou : 0.0f;
      float bce = fmaxf(x, 0.f) - x * t + log1pf(__expf(-fabsf(x)));
      float d = t - p;
      s_cls += d * d * bce;
    }
  }
  __shared__ float rA[256], rB[256];
  __shared__ unsigned rC[256], rD[256];
  const int tid = threadIdx.x;
  rA[tid] = s_cls; rB[tid] = s_box; rC[tid] = c_pos; rD[tid] = c_match;
  __syncthreads();
  for (int s = 128; s > 0; s >>= 1) {
    if (tid < s) { rA[tid] += rA[tid + s]; rB[tid] += rB[tid + s];
                   rC[tid] += rC[tid + s]; rD[tid] += rD[tid + s]; }
    __syncthreads();
  }
  if (tid == 0) {
    atomicAdd(acc + 0, rA[0]); atomicAdd(acc + 1, rB[0]);
    atomicAdd(cnt + 0, rC[0]); atomicAdd(cnt + 1, rD[0]);
  }
}

// ---------------------------------------------------------------- kernel 4
__global__ void k_final(const float* __restrict__ acc,
                        const unsigned* __restrict__ cnt,
                        float* __restrict__ out) {
  if (threadIdx.x == 0 && blockIdx.x == 0) {
    float npc = (float)(cnt[0] > 1u ? cnt[0] : 1u);
    float npm = (float)(cnt[1] > 1u ? cnt[1] : 1u);
    out[0] = acc[0] / npc + 2.0f * acc[1] / npm;
  }
}

// ---------------------------------------------------------------- launcher
extern "C" void kernel_launch(void* const* d_in, const int* in_sizes, int n_in,
                              void* d_out, int out_size, void* d_ws, size_t ws_size,
                              hipStream_t stream) {
  const float*  logits  = (const float*)d_in[0];
  const float4* pboxes  = (const float4*)d_in[1];
  const float*  pointsf = (const float*)d_in[2];
  const float2* points2 = (const float2*)d_in[2];
  const float*  strides = (const float*)d_in[3];
  const float4* gts     = (const float4*)d_in[4];
  const int*    glab    = (const int*)d_in[5];
  float* out = (float*)d_out;

  const int N = in_sizes[2] / 2;
  const int M = in_sizes[5];

  char* ws = (char*)d_ws;
  float*    acc = (float*)ws;                 // [0]=s_cls [1]=s_box
  unsigned* cnt = (unsigned*)(ws + 8);        // [0]=n_pos_cls [1]=n_matched
  float* all_neg = (float*)(ws + 64);
  int*   cand    = (int*)(ws + 64 + (size_t)4 * N);
  size_t off3 = (64 + (size_t)8 * N + 7) & ~(size_t)7;
  unsigned long long* best = (unsigned long long*)(ws + off3);

  k_init<<<(N + 255) / 256, 256, 0, stream>>>(best, acc, cnt, N);
  k_prep<<<(N + 127) / 128, 256, 0, stream>>>(logits, pointsf, strides, gts, N, M,
                                              all_neg, cand);
  const size_t smem = (size_t)8 * N;          // cost + iou columns in LDS
  hipFuncSetAttribute(reinterpret_cast<const void*>(k_cols),
                      hipFuncAttributeMaxDynamicSharedMemorySize, (int)smem);
  k_cols<<<M, 1024, smem, stream>>>(logits, pboxes, points2, strides, gts, glab,
                                    all_neg, cand, best, N);
  k_loss<<<(N + 255) / 256, 256, 0, stream>>>(logits, pboxes, gts, glab, best, N,
                                              acc, cnt);
  k_final<<<1, 1, 0, stream>>>(acc, cnt, out);
}
